// QSDAv2Block_17549236371762
// MI455X (gfx1250) — matrix operand order repaired and checked by reading, not compiled, not run on hardware
//
#include <hip/hip_runtime.h>

typedef __attribute__((ext_vector_type(16))) _Float16 v16h;
typedef __attribute__((ext_vector_type(8)))  _Float16 v8h;
typedef __attribute__((ext_vector_type(8)))  float    v8f;

#define TOK    8192
#define DMODEL 768
#define NHEAD  8
#define DHEAD  32
#define NSTATE 64
#define NSEQ   2048
#define NBATCH 4

// ---------------------------------------------------------------------------
// WMMA GEMM, f16 operands, f32 accumulate. K must be a multiple of 32.
//   C(MxN,f32)[opt] / C16(MxN,f16)[opt] = act(alpha*A*B^T + bias)*rowscale + resid
//   A: MxK f16 row-major (lda); B: NxK f16 row-major (ldb)
//   batched via blockIdx.z (element strides sA/sB/sC/sBias).
// Macro tile 128x64, 8 waves (4x2), each wave 32x32 via 4x v_wmma_f32_16x16x32_f16.
// Tiles staged with double-buffered GLOBAL_LOAD_ASYNC_TO_LDS_B128:
// tile i+1's DMAs are issued before computing tile i; s_wait_asynccnt 3 overlaps
// DMA with WMMA (3 async copies per thread per tile, done-in-order semantics).
// ---------------------------------------------------------------------------
__launch_bounds__(256)
__global__ void qsda_gemm16(
    const _Float16* __restrict__ A, int lda, long long sA,
    const _Float16* __restrict__ B, int ldb, long long sB,
    float* __restrict__ C, int ldc, long long sC,
    _Float16* __restrict__ C16,
    const float* __restrict__ bias, int sBias,
    const float* __restrict__ resid,
    const float* __restrict__ rowscale, int rsStride,
    int M, int N, int Kd, float alpha, int act)
{
  __shared__ _Float16 As[2][128][40];   // [buf][m][k], 16B pad -> 80B row stride
  __shared__ _Float16 Bs[2][64][40];    // [buf][n][k]
  const int bz = blockIdx.z;
  const _Float16* Ab = A + (long long)bz * sA;
  const _Float16* Bb = B + (long long)bz * sB;
  const float* biasb = bias ? (bias + (long long)bz * sBias) : (const float*)0;

  const int tid  = threadIdx.x;
  const int lane = tid & 31;
  const int wave = tid >> 5;
  const int wm = wave >> 1;            // 0..3 -> 32 rows each
  const int wn = wave & 1;             // 0..1 -> 32 cols each
  const int m0 = blockIdx.y * 128;
  const int n0 = blockIdx.x * 64;

  // per-thread DMA segment coordinates (16B per copy, 3 copies per tile)
  const int arow = tid >> 2;           // A rows arow and arow+64
  const int kcs  = (tid & 3) * 8;      // k offset in halves (0/8/16/24)
  long long ga0 = m0 + arow;        if (ga0 >= M) ga0 = M - 1;
  long long ga1 = m0 + arow + 64;   if (ga1 >= M) ga1 = M - 1;
  long long gbn = n0 + arow;        if (gbn >= N) gbn = N - 1;
  const _Float16* gpa0 = Ab + ga0 * lda + kcs;
  const _Float16* gpa1 = Ab + ga1 * lda + kcs;
  const _Float16* gpb  = Bb + gbn * ldb + kcs;

  auto issue = [&](int kb, int buf) {
    unsigned la0 = (unsigned)(unsigned long long)(const void*)&As[buf][arow][kcs];
    unsigned la1 = (unsigned)(unsigned long long)(const void*)&As[buf][arow + 64][kcs];
    unsigned lb  = (unsigned)(unsigned long long)(const void*)&Bs[buf][arow][kcs];
    asm volatile("global_load_async_to_lds_b128 %0, %1, off"
                 :: "v"(la0), "v"((unsigned long long)(const void*)(gpa0 + kb)) : "memory");
    asm volatile("global_load_async_to_lds_b128 %0, %1, off"
                 :: "v"(la1), "v"((unsigned long long)(const void*)(gpa1 + kb)) : "memory");
    asm volatile("global_load_async_to_lds_b128 %0, %1, off"
                 :: "v"(lb),  "v"((unsigned long long)(const void*)(gpb  + kb)) : "memory");
  };

  v8f acc[2][2];
#pragma unroll
  for (int a_ = 0; a_ < 2; ++a_)
#pragma unroll
    for (int b_ = 0; b_ < 2; ++b_)
#pragma unroll
      for (int i = 0; i < 8; ++i) acc[a_][b_][i] = 0.f;

  const int T = Kd >> 5;               // K tiles (K % 32 == 0 by construction)
  issue(0, 0);
  for (int i = 0; i < T; ++i) {
    const int buf = i & 1;
    if (i + 1 < T) {
      issue((i + 1) * 32, buf ^ 1);
      asm volatile("s_wait_asynccnt 0x3" ::: "memory");   // tile i landed; i+1 in flight
    } else {
      asm volatile("s_wait_asynccnt 0x0" ::: "memory");
    }
    __syncthreads();                   // all waves' tile-i DMAs complete

    v16h af[2];
#pragma unroll
    for (int mi = 0; mi < 2; ++mi) {
      int r = wm * 32 + mi * 16 + (lane & 15);
      int kh = (lane < 16) ? 0 : 8;    // A frag: runs [kh,kh+8) and [kh+16,kh+24)
      v8h lo = *(const v8h*)&As[buf][r][kh];
      v8h hi = *(const v8h*)&As[buf][r][kh + 16];
      af[mi] = __builtin_shufflevector(lo, hi, 0,1,2,3,4,5,6,7,8,9,10,11,12,13,14,15);
    }
    v16h bf[2];
#pragma unroll
    for (int ni = 0; ni < 2; ++ni) {
      int n = wn * 32 + ni * 16 + (lane & 15);
      int kh = (lane < 16) ? 0 : 8;    // B frag: runs [kh,kh+8) and [kh+16,kh+24)
      v8h lo = *(const v8h*)&Bs[buf][n][kh];
      v8h hi = *(const v8h*)&Bs[buf][n][kh + 16];
      bf[ni] = __builtin_shufflevector(lo, hi, 0,1,2,3,4,5,6,7,8,9,10,11,12,13,14,15);
    }
#pragma unroll
    for (int mi = 0; mi < 2; ++mi)
#pragma unroll
      for (int ni = 0; ni < 2; ++ni)
        acc[mi][ni] = __builtin_amdgcn_wmma_f32_16x16x32_f16(
            false, af[mi], false, bf[ni], (short)0, acc[mi][ni], false, false);
    __syncthreads();                   // safe to overwrite buf at iteration i+2
  }

  // ---- epilogue ----
#pragma unroll
  for (int mi = 0; mi < 2; ++mi) {
    const int rbase = m0 + wm * 32 + mi * 16 + ((lane < 16) ? 0 : 8);
#pragma unroll
    for (int ni = 0; ni < 2; ++ni) {
      const int col = n0 + wn * 32 + ni * 16 + (lane & 15);
#pragma unroll
      for (int j = 0; j < 8; ++j) {
        int row = rbase + j;
        if (row < M && col < N) {
          float x = alpha * acc[mi][ni][j];
          if (biasb) x += biasb[col];
          if (act == 1)      x = x / (1.f + __expf(-x));                            // SiLU
          else if (act == 2) x = 0.5f * x * (1.f + erff(x * 0.7071067811865475f));  // GELU
          if (rowscale) x *= rowscale[(long long)row * rsStride];
          if (resid)    x += resid[(long long)row * ldc + col];
          long long o = (long long)bz * sC + (long long)row * ldc + col;
          if (C)   C[o]   = x;
          if (C16) C16[o] = (_Float16)x;
        }
      }
    }
  }
}

// ---------------------------------------------------------------------------
// f32 -> f16 convert (grid-stride)
__global__ void qsda_cvt(const float* __restrict__ src, _Float16* __restrict__ dst,
                         long long n)
{
  long long i = (long long)blockIdx.x * blockDim.x + threadIdx.x;
  long long stride = (long long)gridDim.x * blockDim.x;
  for (; i < n; i += stride) dst[i] = (_Float16)src[i];
}

// f32 (rows x cols) -> f16 (rows x ldd) with zero padding of cols..ldd-1
__global__ void qsda_cvt_pad(const float* __restrict__ src, _Float16* __restrict__ dst,
                             int rows, int cols, int ldd)
{
  int i = blockIdx.x * blockDim.x + threadIdx.x;
  if (i >= rows * ldd) return;
  int r = i / ldd, c = i - r * ldd;
  dst[i] = (c < cols) ? (_Float16)src[r * cols + c] : (_Float16)0.f;
}

// zero-fill f16 buffer
__global__ void qsda_fillh0(_Float16* __restrict__ dst, long long n)
{
  long long i = (long long)blockIdx.x * blockDim.x + threadIdx.x;
  long long stride = (long long)gridDim.x * blockDim.x;
  for (; i < n; i += stride) dst[i] = (_Float16)0.f;
}

// batched transpose + convert: src[b] is RxC f32, dst[b] is CxR f16
__global__ void qsda_tcvt(const float* __restrict__ src, _Float16* __restrict__ dst,
                          int R, int Ccols, long long total)
{
  long long i = (long long)blockIdx.x * blockDim.x + threadIdx.x;
  if (i >= total) return;
  long long per = (long long)R * Ccols;
  long long b = i / per;
  long long rem = i - b * per;
  int r = (int)(rem / Ccols);
  int c = (int)(rem - (long long)r * Ccols);
  dst[b * per + (long long)c * R + r] = (_Float16)src[i];
}

// ---------------------------------------------------------------------------
// Joint L2-normalize complex rows of length 32 (one wave32 per row)
__global__ void qsda_norm32(const float* __restrict__ re_in, const float* __restrict__ im_in,
                            float* __restrict__ re_out, float* __restrict__ im_out, int rows)
{
  int row = blockIdx.x;
  if (row >= rows) return;
  int l = threadIdx.x;
  float r = re_in[(long long)row * 32 + l];
  float i = im_in[(long long)row * 32 + l];
  float s = r * r + i * i;
  for (int off = 16; off; off >>= 1) s += __shfl_xor(s, off, 32);
  float n = sqrtf(s);
  n = fmaxf(n, 1e-12f);
  re_out[(long long)row * 32 + l] = r / n;
  im_out[(long long)row * 32 + l] = i / n;
}

// p[h,bn] = 0.95 * sigmoid(z[h,bn,:] . pw2[h] + pb2[h])
__global__ void qsda_p_from_z(const float* __restrict__ z, const float* __restrict__ pw2,
                              const float* __restrict__ pb2, float* __restrict__ p)
{
  int idx = blockIdx.x * blockDim.x + threadIdx.x;
  if (idx >= NHEAD * TOK) return;
  int hh = idx >> 13;
  const float* zr = z + (long long)idx * 16;
  float acc = pb2[hh];
#pragma unroll
  for (int k = 0; k < 16; ++k) acc += zr[k] * pw2[hh * 16 + k];
  p[idx] = 0.95f / (1.f + __expf(-acc));
}

// G[h'][h*64+s] = sum_D values[h,s,D] * gate_w[h', h*768+D]
__global__ void qsda_G(const float* __restrict__ values, const float* __restrict__ gate_w,
                       float* __restrict__ G)
{
  int idx = blockIdx.x * blockDim.x + threadIdx.x;
  if (idx >= NHEAD * NHEAD * NSTATE) return;
  int hp = idx >> 9;
  int hs = idx & 511;
  int hh = hs >> 6;
  int ss = hs & 63;
  const float* vp = values + ((long long)hh * NSTATE + ss) * DMODEL;
  const float* gw = gate_w + (long long)hp * (NHEAD * DMODEL) + (long long)hh * DMODEL;
  float acc = 0.f;
  for (int d0 = 0; d0 < DMODEL; ++d0) acc += vp[d0] * gw[d0];
  G[idx] = acc;
}

// mixed-state measurement attention: one block per (token, head), 64 threads
__global__ void qsda_attn(const float* __restrict__ pr, const float* __restrict__ pi,
                          const float* __restrict__ mr, const float* __restrict__ mi,
                          const float* __restrict__ p, float* __restrict__ attn)
{
  __shared__ float prs[32], pis[32], rawv[64], ssum;
  int bn = blockIdx.x, hh = blockIdx.y, s = threadIdx.x;
  long long base = ((long long)hh * TOK + bn) * 32;
  if (s < 32) prs[s] = pr[base + s];
  else        pis[s - 32] = pi[base + s - 32];
  __syncthreads();
  float ore = 0.f, oim = 0.f;
  const float* mrp = mr + ((long long)hh * NSTATE + s) * 32;
  const float* mip = mi + ((long long)hh * NSTATE + s) * 32;
#pragma unroll
  for (int d0 = 0; d0 < 32; ++d0) {
    ore += prs[d0] * mrp[d0] + pis[d0] * mip[d0];
    oim += pis[d0] * mrp[d0] - prs[d0] * mip[d0];
  }
  float pv = p[(long long)hh * TOK + bn];
  float raw = (1.f - pv) * (ore * ore + oim * oim) + pv * (1.f / 32.f);
  rawv[s] = raw;
  __syncthreads();
  if (s == 0) {
    float t = 0.f;
    for (int i = 0; i < 64; ++i) t += rawv[i];
    ssum = t;
  }
  __syncthreads();
  attn[(long long)bn * 512 + hh * 64 + s] = raw / (ssum + 1e-8f);
}

// gate logits via precomputed G, softmax over 8 heads
__global__ void qsda_gates(const float* __restrict__ attn, const float* __restrict__ G,
                           const float* __restrict__ gate_b, float* __restrict__ gates)
{
  __shared__ float arow[512];
  __shared__ float gl[8];
  int bn = blockIdx.x, t = threadIdx.x;
#pragma unroll
  for (int j = 0; j < 8; ++j) arow[t + j * 64] = attn[(long long)bn * 512 + t + j * 64];
  __syncthreads();
  if (t < 8) {
    float acc = gate_b[t];
    const float* Gp = G + t * 512;
    for (int i = 0; i < 512; ++i) acc += arow[i] * Gp[i];
    gl[t] = acc;
  }
  __syncthreads();
  if (t == 0) {
    float mx = gl[0];
    for (int i = 1; i < 8; ++i) mx = fmaxf(mx, gl[i]);
    float se = 0.f, e[8];
    for (int i = 0; i < 8; ++i) { e[i] = __expf(gl[i] - mx); se += e[i]; }
    for (int i = 0; i < 8; ++i) gates[(long long)bn * 8 + i] = e[i] / se;
  }
}

// wattn (f16, feeds GEMM A) = attn * gate
__global__ void qsda_wattn(const float* __restrict__ attn, const float* __restrict__ gates,
                           _Float16* __restrict__ wattn)
{
  long long idx = (long long)blockIdx.x * blockDim.x + threadIdx.x;
  if (idx >= (long long)TOK * 512) return;
  long long bn = idx >> 9;
  int hs = (int)(idx & 511);
  wattn[idx] = (_Float16)(attn[idx] * gates[bn * 8 + (hs >> 6)]);
}

__global__ void qsda_ptok(const float* __restrict__ p, float* __restrict__ ptok,
                          float* __restrict__ om)
{
  int bn = blockIdx.x * blockDim.x + threadIdx.x;
  if (bn >= TOK) return;
  float s = 0.f;
#pragma unroll
  for (int hh = 0; hh < NHEAD; ++hh) s += p[(long long)hh * TOK + bn];
  s *= (1.f / NHEAD);
  ptok[bn] = s;
  om[bn] = 1.f - s;
}

// fused row softmax of coupling + neighbor dot + p_new; stores (1 - p_new)
__global__ void qsda_couple(const float* __restrict__ coup, const float* __restrict__ om,
                            const float* __restrict__ ptok, const float* __restrict__ beta,
                            float* __restrict__ ompnew)
{
  __shared__ float red[256];
  int b = blockIdx.y, n = blockIdx.x, t = threadIdx.x;
  const float* rowp = coup + ((long long)(b * NSEQ + n)) * NSEQ;
  const float* omb = om + (long long)b * NSEQ;
  float mx = -3.4e38f;
  for (int m = t; m < NSEQ; m += 256) mx = fmaxf(mx, rowp[m]);
  red[t] = mx; __syncthreads();
  for (int off = 128; off; off >>= 1) { if (t < off) red[t] = fmaxf(red[t], red[t + off]); __syncthreads(); }
  mx = red[0]; __syncthreads();
  float se = 0.f, sw = 0.f;
  for (int m = t; m < NSEQ; m += 256) {
    float e = __expf(rowp[m] - mx);
    se += e; sw += e * omb[m];
  }
  red[t] = se; __syncthreads();
  for (int off = 128; off; off >>= 1) { if (t < off) red[t] += red[t + off]; __syncthreads(); }
  se = red[0]; __syncthreads();
  red[t] = sw; __syncthreads();
  for (int off = 128; off; off >>= 1) { if (t < off) red[t] += red[t + off]; __syncthreads(); }
  if (t == 0) {
    float neigh = red[0] / se;
    float pt = ptok[b * NSEQ + n];
    float sb = 1.f / (1.f + __expf(-beta[0]));
    float pn = pt * (1.f - sb * neigh);
    pn = fminf(fmaxf(pn, 0.f), 0.95f);
    ompnew[b * NSEQ + n] = 1.f - pn;
  }
}

// <p_s|c_s>
__global__ void qsda_gpc(const float* __restrict__ Pnr, const float* __restrict__ Pni,
                         const float* __restrict__ Cnr, const float* __restrict__ Cni,
                         float* __restrict__ gre, float* __restrict__ gim)
{
  int s = threadIdx.x;
  float gr = 0.f, gi = 0.f;
#pragma unroll
  for (int d0 = 0; d0 < 32; ++d0) {
    float a = Pnr[s * 32 + d0], bb = Pni[s * 32 + d0];
    float c = Cnr[s * 32 + d0], dd = Cni[s * 32 + d0];
    gr += a * c + bb * dd;
    gi += a * dd - bb * c;
  }
  gre[s] = gr; gim[s] = gi;
}

// interference features (f16, feed GEMM A): one wave per token
__global__ void qsda_interf(const float* __restrict__ qr, const float* __restrict__ qi,
                            const float* __restrict__ Pnr, const float* __restrict__ Pni,
                            const float* __restrict__ Cnr, const float* __restrict__ Cni,
                            const float* __restrict__ gre, const float* __restrict__ gim,
                            _Float16* __restrict__ feats)
{
  __shared__ float q_r[32], q_i[32];
  int bn = blockIdx.x, s = threadIdx.x;
  q_r[s] = qr[(long long)bn * 32 + s];
  q_i[s] = qi[(long long)bn * 32 + s];
  __syncthreads();
  float ar = 0.f, ai = 0.f, br = 0.f, bi = 0.f;
#pragma unroll
  for (int d0 = 0; d0 < 32; ++d0) {
    float qrd = q_r[d0], qid = q_i[d0];
    float Prd = Pnr[s * 32 + d0], Pid = Pni[s * 32 + d0];
    float Crd = Cnr[s * 32 + d0], Cid = Cni[s * 32 + d0];
    ar += qrd * Prd + qid * Pid;
    ai += qrd * Pid - qid * Prd;
    br += qrd * Crd + qid * Cid;
    bi += qrd * Cid - qid * Crd;
  }
  float gr = gre[s], gi = gim[s];
  float tr = ar * gr - ai * gi;
  float ti = ar * gi + ai * gr;
  feats[(long long)bn * 64 + s]      = (_Float16)(tr * br + ti * bi);
  feats[(long long)bn * 64 + 32 + s] = (_Float16)(ti * br - tr * bi);
}

// LayerNorm(768) + affine + exact GELU, writes f16 (feeds GEMM A)
__global__ void qsda_ln_gelu(const float* __restrict__ x, const float* __restrict__ g,
                             const float* __restrict__ bta, _Float16* __restrict__ o16)
{
  __shared__ float red[256];
  int row = blockIdx.x, t = threadIdx.x;
  const float* xp = x + (long long)row * DMODEL;
  float v[3];
  float s = 0.f;
#pragma unroll
  for (int j = 0; j < 3; ++j) { v[j] = xp[t + j * 256]; s += v[j]; }
  red[t] = s; __syncthreads();
  for (int off = 128; off; off >>= 1) { if (t < off) red[t] += red[t + off]; __syncthreads(); }
  float mu = red[0] * (1.f / DMODEL);
  __syncthreads();
  float s2 = 0.f;
#pragma unroll
  for (int j = 0; j < 3; ++j) { float d0 = v[j] - mu; s2 += d0 * d0; }
  red[t] = s2; __syncthreads();
  for (int off = 128; off; off >>= 1) { if (t < off) red[t] += red[t + off]; __syncthreads(); }
  float rstd = rsqrtf(red[0] * (1.f / DMODEL) + 1e-5f);
#pragma unroll
  for (int j = 0; j < 3; ++j) {
    int c = t + j * 256;
    float y = (v[j] - mu) * rstd * g[c] + bta[c];
    y = 0.5f * y * (1.f + erff(y * 0.7071067811865475f));
    o16[(long long)row * DMODEL + c] = (_Float16)y;
  }
}

__global__ void qsda_softmax3(const float* __restrict__ lg, float* __restrict__ rg)
{
  int idx = blockIdx.x * blockDim.x + threadIdx.x;
  if (idx >= TOK) return;
  float a0 = lg[idx * 3], a1 = lg[idx * 3 + 1], a2 = lg[idx * 3 + 2];
  float mx = fmaxf(a0, fmaxf(a1, a2));
  float e0 = __expf(a0 - mx), e1 = __expf(a1 - mx), e2 = __expf(a2 - mx);
  float s = e0 + e1 + e2;
  rg[idx * 3]     = e0 / s;
  rg[idx * 3 + 1] = e1 / s;
  rg[idx * 3 + 2] = e2 / s;
}

// ---------------------------------------------------------------------------
extern "C" void kernel_launch(void* const* d_in, const int* in_sizes, int n_in,
                              void* d_out, int out_size, void* d_ws, size_t ws_size,
                              hipStream_t stream)
{
  const float* h       = (const float*)d_in[0];
  const float* W_real  = (const float*)d_in[1];
  const float* W_imag  = (const float*)d_in[2];
  const float* pw1     = (const float*)d_in[3];
  const float* pb1     = (const float*)d_in[4];
  const float* pw2     = (const float*)d_in[5];
  const float* pb2     = (const float*)d_in[6];
  const float* m_real  = (const float*)d_in[7];
  const float* m_imag  = (const float*)d_in[8];
  const float* values  = (const float*)d_in[9];
  const float* gate_w  = (const float*)d_in[10];
  const float* gate_b  = (const float*)d_in[11];
  const float* out_w   = (const float*)d_in[12];
  const float* out_b   = (const float*)d_in[13];
  const float* beta    = (const float*)d_in[14];
  const float* P_real  = (const float*)d_in[15];
  const float* P_imag  = (const float*)d_in[16];
  const float* C_real  = (const float*)d_in[17];
  const float* C_imag  = (const float*)d_in[18];
  const float* enc_r_w = (const float*)d_in[19];
  const float* enc_i_w = (const float*)d_in[20];
  const float* r_out_w = (const float*)d_in[21];
  const float* r_out_b = (const float*)d_in[22];
  const float* ew1     = (const float*)d_in[23];
  const float* eb1     = (const float*)d_in[24];
  const float* eg      = (const float*)d_in[25];
  const float* ebt     = (const float*)d_in[26];
  const float* ew2     = (const float*)d_in[27];
  const float* eb2     = (const float*)d_in[28];
  const float* rw1     = (const float*)d_in[29];
  const float* rb1     = (const float*)d_in[30];
  const float* rw2     = (const float*)d_in[31];
  const float* rb2     = (const float*)d_in[32];
  float* out = (float*)d_out;
  (void)ws_size; (void)in_sizes; (void)n_in; (void)out_size;

  char* base = (char*)d_ws;
  size_t off = 0;
  auto allocB = [&](size_t bytes) {
    void* p = base + off;
    off = (off + bytes + 255) & ~(size_t)255;
    return p;
  };
  auto allocF = [&](size_t n) { return (float*)allocB(n * 4); };
  auto allocH = [&](size_t n) { return (_Float16*)allocB(n * 2); };

  // f32 intermediates
  float* pr    = allocF((size_t)NHEAD * TOK * DHEAD);
  float* pi    = allocF((size_t)NHEAD * TOK * DHEAD);
  float* mrn   = allocF((size_t)NHEAD * NSTATE * DHEAD);
  float* minv  = allocF((size_t)NHEAD * NSTATE * DHEAD);
  float* z     = allocF((size_t)NHEAD * TOK * 16);
  float* p     = allocF((size_t)NHEAD * TOK);
  float* attn  = allocF((size_t)TOK * 512);
  float* G     = allocF((size_t)NHEAD * NHEAD * NSTATE);
  float* gates = allocF((size_t)TOK * NHEAD);
  float* h1    = allocF((size_t)TOK * DMODEL);
  float* ptok  = allocF((size_t)TOK);
  float* om    = allocF((size_t)TOK);
  float* ompn  = allocF((size_t)TOK);
  float* coup  = allocF((size_t)NBATCH * NSEQ * NSEQ);
  float* qr    = allocF((size_t)TOK * 32);
  float* qi    = allocF((size_t)TOK * 32);
  float* Pnr   = allocF(32 * 32);
  float* Pni   = allocF(32 * 32);
  float* Cnr   = allocF(32 * 32);
  float* Cni   = allocF(32 * 32);
  float* gre   = allocF(32);
  float* gim   = allocF(32);
  float* h2    = allocF((size_t)TOK * DMODEL);
  float* rglog = allocF((size_t)TOK * 3);
  float* rg    = allocF((size_t)TOK * 3);
  float* xk    = allocF((size_t)TOK * DMODEL);

  // f16 operands
  _Float16* hc     = allocH((size_t)TOK * DMODEL);
  _Float16* Wrt    = allocH((size_t)NHEAD * DHEAD * DMODEL);
  _Float16* Wit    = allocH((size_t)NHEAD * DHEAD * DMODEL);
  _Float16* pw1c   = allocH((size_t)NHEAD * 16 * DMODEL);
  _Float16* vtc    = allocH((size_t)DMODEL * 512);
  _Float16* outwc  = allocH((size_t)DMODEL * DMODEL);
  _Float16* wattc  = allocH((size_t)TOK * 512);
  _Float16* combc  = allocH((size_t)TOK * DMODEL);
  _Float16* h1c    = allocH((size_t)TOK * DMODEL);
  _Float16* encrc  = allocH((size_t)32 * DMODEL);
  _Float16* encic  = allocH((size_t)32 * DMODEL);
  _Float16* featsc = allocH((size_t)TOK * 64);
  _Float16* routwc = allocH((size_t)DMODEL * 64);
  _Float16* h2c    = allocH((size_t)TOK * DMODEL);
  _Float16* rw1c   = allocH((size_t)16 * DMODEL);
  _Float16* g1c    = allocH((size_t)TOK * 32);       // padded K: 16 real + 16 zero
  _Float16* rw2p   = allocH((size_t)3 * 32);         // padded K
  _Float16* ew1c   = allocH((size_t)3 * DMODEL * DMODEL);
  _Float16* ew2c   = allocH((size_t)3 * DMODEL * DMODEL);
  _Float16* xkc    = allocH((size_t)TOK * DMODEL);

  auto cvt = [&](const float* s, _Float16* d, long long n) {
    int blocks = (int)((n + 1023) / 1024); if (blocks > 4096) blocks = 4096;
    qsda_cvt<<<blocks, 256, 0, stream>>>(s, d, n);
  };
  auto tcvt = [&](const float* s, _Float16* d, int R, int Cc, int batches) {
    long long total = (long long)R * Cc * batches;
    qsda_tcvt<<<(int)((total + 255) / 256), 256, 0, stream>>>(s, d, R, Cc, total);
  };
  auto gemm = [&](const _Float16* Am, int lda, long long sA,
                  const _Float16* Bm, int ldb, long long sB,
                  float* Cm, _Float16* C16, int ldc, long long sC,
                  const float* bias, int sBias,
                  const float* resid, const float* rowscale, int rsStride,
                  int M, int N, int K, float alpha, int act, int batches) {
    dim3 g((N + 63) / 64, (M + 127) / 128, batches);
    qsda_gemm16<<<g, 256, 0, stream>>>(Am, lda, sA, Bm, ldb, sB, Cm, ldc, sC, C16,
                                       bias, sBias, resid, rowscale, rsStride,
                                       M, N, K, alpha, act);
  };

  // ---- one-time f16 conversions / weight transposes ----
  cvt(h, hc, (long long)TOK * DMODEL);
  tcvt(W_real, Wrt, DMODEL, DHEAD, NHEAD);         // (768x32) -> (32x768) per head
  tcvt(W_imag, Wit, DMODEL, DHEAD, NHEAD);
  cvt(pw1, pw1c, (long long)NHEAD * 16 * DMODEL);
  tcvt(values, vtc, 512, DMODEL, 1);               // (512x768) -> (768x512)
  cvt(out_w, outwc, (long long)DMODEL * DMODEL);
  cvt(enc_r_w, encrc, 32 * DMODEL);
  cvt(enc_i_w, encic, 32 * DMODEL);
  cvt(r_out_w, routwc, (long long)DMODEL * 64);
  cvt(rw1, rw1c, 16 * DMODEL);
  qsda_cvt_pad<<<1, 96, 0, stream>>>(rw2, rw2p, 3, 16, 32);
  cvt(ew1, ew1c, (long long)3 * DMODEL * DMODEL);
  cvt(ew2, ew2c, (long long)3 * DMODEL * DMODEL);

  // ---- MultiHeadQuantumAttention ----
  gemm(hc, DMODEL, 0, Wrt, DMODEL, (long long)DHEAD * DMODEL,
       pr, nullptr, DHEAD, (long long)TOK * DHEAD, nullptr, 0, nullptr, nullptr, 0,
       TOK, DHEAD, DMODEL, 1.f, 0, NHEAD);
  gemm(hc, DMODEL, 0, Wit, DMODEL, (long long)DHEAD * DMODEL,
       pi, nullptr, DHEAD, (long long)TOK * DHEAD, nullptr, 0, nullptr, nullptr, 0,
       TOK, DHEAD, DMODEL, 1.f, 0, NHEAD);
  qsda_norm32<<<NHEAD * TOK, 32, 0, stream>>>(pr, pi, pr, pi, NHEAD * TOK);

  gemm(hc, DMODEL, 0, pw1c, DMODEL, (long long)16 * DMODEL,
       z, nullptr, 16, (long long)TOK * 16, pb1, 16, nullptr, nullptr, 0,
       TOK, 16, DMODEL, 1.f, 1, NHEAD);                              // SiLU
  qsda_p_from_z<<<(NHEAD * TOK + 255) / 256, 256, 0, stream>>>(z, pw2, pb2, p);

  qsda_norm32<<<NHEAD * NSTATE, 32, 0, stream>>>(m_real, m_imag, mrn, minv, NHEAD * NSTATE);
  qsda_attn<<<dim3(TOK, NHEAD), 64, 0, stream>>>(pr, pi, mrn, minv, p, attn);

  qsda_G<<<(NHEAD * NHEAD * NSTATE + 255) / 256, 256, 0, stream>>>(values, gate_w, G);
  qsda_gates<<<TOK, 64, 0, stream>>>(attn, G, gate_b, gates);
  qsda_wattn<<<(TOK * 512 + 255) / 256, 256, 0, stream>>>(attn, gates, wattc);

  gemm(wattc, 512, 0, vtc, 512, 0, nullptr, combc, DMODEL, 0,
       nullptr, 0, nullptr, nullptr, 0, TOK, DMODEL, 512, 1.f, 0, 1);
  gemm(combc, DMODEL, 0, outwc, DMODEL, 0, h1, h1c, DMODEL, 0,
       out_b, 0, h, nullptr, 0, TOK, DMODEL, DMODEL, 1.f, 0, 1);     // + residual h
  qsda_ptok<<<(TOK + 255) / 256, 256, 0, stream>>>(p, ptok, om);

  // ---- EntanglementPropagationLayer ----
  gemm(h1c, DMODEL, (long long)NSEQ * DMODEL, h1c, DMODEL, (long long)NSEQ * DMODEL,
       coup, nullptr, NSEQ, (long long)NSEQ * NSEQ, nullptr, 0, nullptr, nullptr, 0,
       NSEQ, NSEQ, DMODEL, 0.03608439182435161f, 0, NBATCH);
  qsda_couple<<<dim3(NSEQ, NBATCH), 256, 0, stream>>>(coup, om, ptok, beta, ompn);

  // ---- QuantumInterferenceReasoning ----
  gemm(h1c, DMODEL, 0, encrc, DMODEL, 0, qr, nullptr, 32, 0,
       nullptr, 0, nullptr, nullptr, 0, TOK, 32, DMODEL, 1.f, 0, 1);
  gemm(h1c, DMODEL, 0, encic, DMODEL, 0, qi, nullptr, 32, 0,
       nullptr, 0, nullptr, nullptr, 0, TOK, 32, DMODEL, 1.f, 0, 1);
  qsda_norm32<<<TOK, 32, 0, stream>>>(qr, qi, qr, qi, TOK);
  qsda_norm32<<<32, 32, 0, stream>>>(P_real, P_imag, Pnr, Pni, 32);
  qsda_norm32<<<32, 32, 0, stream>>>(C_real, C_imag, Cnr, Cni, 32);
  qsda_gpc<<<1, 32, 0, stream>>>(Pnr, Pni, Cnr, Cni, gre, gim);
  qsda_interf<<<TOK, 32, 0, stream>>>(qr, qi, Pnr, Pni, Cnr, Cni, gre, gim, featsc);
  gemm(featsc, 64, 0, routwc, 64, 0, h2, h2c, DMODEL, 0,
       r_out_b, 0, h1, ompn, 1, TOK, DMODEL, 64, 1.f, 0, 1);         // h2 = h1+(1-p_new)*reason

  // ---- AdaptiveHilbertRouter ----
  qsda_fillh0<<<512, 256, 0, stream>>>(g1c, (long long)TOK * 32);    // zero K-padding
  gemm(h2c, DMODEL, 0, rw1c, DMODEL, 0, nullptr, g1c, 32, 0,
       rb1, 0, nullptr, nullptr, 0, TOK, 16, DMODEL, 1.f, 2, 1);     // GELU, ldc=32 (padded)
  gemm(g1c, 32, 0, rw2p, 32, 0, rglog, nullptr, 3, 0,
       rb2, 0, nullptr, nullptr, 0, TOK, 3, 32, 1.f, 0, 1);          // K padded to 32
  qsda_softmax3<<<(TOK + 255) / 256, 256, 0, stream>>>(rglog, rg);

  hipMemcpyAsync(out, h2, (size_t)TOK * DMODEL * sizeof(float),
                 hipMemcpyDeviceToDevice, stream);
  for (int k = 0; k < 3; ++k) {
    gemm(h2c, DMODEL, 0, ew1c + (size_t)k * DMODEL * DMODEL, DMODEL, 0,
         xk, nullptr, DMODEL, 0, eb1 + (size_t)k * DMODEL, 0, nullptr, nullptr, 0,
         TOK, DMODEL, DMODEL, 1.f, 0, 1);
    qsda_ln_gelu<<<TOK, 256, 0, stream>>>(xk, eg + (size_t)k * DMODEL,
                                          ebt + (size_t)k * DMODEL, xkc);
    gemm(xkc, DMODEL, 0, ew2c + (size_t)k * DMODEL * DMODEL, DMODEL, 0,
         out, nullptr, DMODEL, 0, eb2 + (size_t)k * DMODEL, 0, out, rg + k, 3,
         TOK, DMODEL, DMODEL, 1.f, 0, 1);                            // out += rg_k*y_k
  }
}
